// ErdosNet_1640677507203
// MI455X (gfx1250) — compile-verified, matrix operand order checked by reading, not run on hardware
//
#include <hip/hip_runtime.h>
#include <hip/hip_bf16.h>

#define N_NODES 50000
#define N_EDGES 800000
#define DIM 128
#define HID 128
#define NGRAPH 64
#define NEG 0.01f

typedef __attribute__((ext_vector_type(16))) __bf16 bf16x16;
typedef __attribute__((ext_vector_type(8)))  float  f32x8;

__device__ __forceinline__ f32x8 wmma_bf16(bf16x16 a, bf16x16 b, f32x8 c) {
  // D = A(16x32 bf16) * B(32x16 bf16) + C(16x16 f32)
  return __builtin_amdgcn_wmma_f32_16x16x32_bf16(false, a, false, b, (short)0, c, false, false);
}

// A-fragment (16x32 tile of a row-major fp32 [16 x ld] stripe), converted to bf16.
// Lane l: row m=l&15; lanes<16 hold K {0..7,16..23}, lanes>=16 hold K {8..15,24..31} (+32*kc).
__device__ __forceinline__ bf16x16 load_a_frag(const float* __restrict__ base, int ld, int kc, int lane) {
  int m  = lane & 15;
  int kb = (lane & 16) ? 8 : 0;
  const float* p = base + m * ld + kc * 32 + kb;
  bf16x16 a;
#pragma unroll
  for (int i = 0; i < 8; ++i) {
    a[i]     = (__bf16)p[i];
    a[i + 8] = (__bf16)p[i + 16];
  }
  return a;
}

// B-fragment from pre-packed layout: Bp[(((ct*4)+kc)*32 + lane)*16 + i] -> one 32B load.
__device__ __forceinline__ bf16x16 load_b_frag(const __bf16* __restrict__ Bp, int ct, int kc, int lane) {
  return *(const bf16x16*)(Bp + ((((ct * 4 + kc) * 32) + lane) << 4));
}

__device__ __forceinline__ float leaky(float v)    { return v > 0.f ? v : NEG * v; }
__device__ __forceinline__ float sigmoidf_(float v){ return 1.f / (1.f + __expf(-v)); }

// ---------- weight packing into WMMA B-fragment layout (bf16) ----------
// !transposed: src(k,n) = W[k*Kout + n]   (W is [128 x Kout] row-major)
//  transposed: src(k,n) = W[n*128  + k]   (W is [Kout x 128], we need W^T)
__global__ void k_pack(const float* __restrict__ W, __bf16* __restrict__ Bp, int Kout, int transposed) {
  int tid = blockIdx.x * 256 + threadIdx.x;
  if (tid >= Kout * 128) return;
  int i    = tid & 15;
  int lane = (tid >> 4) & 31;
  int c    = (tid >> 9) & 3;
  int ct   = tid >> 11;
  int n  = ct * 16 + (lane & 15);
  int kb = (lane & 16) ? 8 : 0;
  int k  = c * 32 + kb + (i < 8 ? i : i + 8);
  float v = transposed ? W[n * 128 + k] : W[k * Kout + n];
  Bp[tid] = (__bf16)v;
}

__global__ void k_zero(float* __restrict__ p, int n) {
  int i = blockIdx.x * 256 + threadIdx.x;
  if (i < n) p[i] = 0.f;
}

__global__ void k_degree(const int* __restrict__ row, const int* __restrict__ col,
                         float* __restrict__ degC, float* __restrict__ degR) {
  int e = blockIdx.x * 256 + threadIdx.x;
  if (e >= N_EDGES) return;
  atomicAdd(&degC[col[e]], 1.f);
  atomicAdd(&degR[row[e]], 1.f);
}

__global__ void k_dinv(const float* __restrict__ degC, float* __restrict__ dinv) {
  int i = blockIdx.x * 256 + threadIdx.x;
  if (i < N_NODES) dinv[i] = rsqrtf(degC[i] + 1.f);   // +1 self loop
}

// ---------- WMMA GEMM, Kout fixed at 128: one wave = one 16x16 tile, no loop ----------
__global__ __launch_bounds__(256) void k_gemm(const float* __restrict__ A, const __bf16* __restrict__ Bp,
                                              const float* __restrict__ bias, float* __restrict__ C,
                                              int act) {
  int tile = blockIdx.x;
  int lane = threadIdx.x & 31;
  int ct   = threadIdx.x >> 5;               // 8 waves = 8 column tiles of 128
  const float* Ab = A + (size_t)tile * 16 * DIM;

  bf16x16 a0 = load_a_frag(Ab, DIM, 0, lane);
  bf16x16 a1 = load_a_frag(Ab, DIM, 1, lane);
  bf16x16 a2 = load_a_frag(Ab, DIM, 2, lane);
  bf16x16 a3 = load_a_frag(Ab, DIM, 3, lane);
  bf16x16 b0 = load_b_frag(Bp, ct, 0, lane);
  bf16x16 b1 = load_b_frag(Bp, ct, 1, lane);
  bf16x16 b2 = load_b_frag(Bp, ct, 2, lane);
  bf16x16 b3 = load_b_frag(Bp, ct, 3, lane);

  // Two independent accumulators: dependent WMMAs are 2 apart -> hazard slots filled.
  f32x8 acc0 = {}, acc1 = {};
  acc0 = wmma_bf16(a0, b0, acc0);
  acc1 = wmma_bf16(a1, b1, acc1);
  acc0 = wmma_bf16(a2, b2, acc0);
  acc1 = wmma_bf16(a3, b3, acc1);

  float bv = bias ? bias[ct * 16 + (lane & 15)] : 0.f;
  int n    = ct * 16 + (lane & 15);
  float* Cb = C + (size_t)(tile * 16 + ((lane & 16) ? 8 : 0)) * 128 + n;
#pragma unroll
  for (int r = 0; r < 8; ++r) {              // constant 512B row stride -> imm offsets
    float v = acc0[r] + acc1[r] + bv;
    if (act == 1) v = leaky(v);
    Cb[r * 128] = v;
  }
}

// ---------- edge scatter-add: dst[col] += src[row] * (norm?) ; wave32 per edge ----------
__global__ __launch_bounds__(256) void k_scatter(const float* __restrict__ src, const int* __restrict__ row,
                                                 const int* __restrict__ col, const float* __restrict__ dinv,
                                                 float* __restrict__ dst, int useNorm) {
  int e = blockIdx.x * 8 + (threadIdx.x >> 5);
  if (e >= N_EDGES) return;
  int lane = threadIdx.x & 31;
  int r = row[e], c = col[e];
  float s = useNorm ? dinv[r] * dinv[c] : 1.f;
  float4 v = *(const float4*)(src + (size_t)r * HID + lane * 4);
  float* d = dst + (size_t)c * HID + lane * 4;
  atomicAdd(d + 0, v.x * s);
  atomicAdd(d + 1, v.y * s);
  atomicAdd(d + 2, v.z * s);
  atomicAdd(d + 3, v.w * s);
}

__global__ void k_gcn_finish(const float* __restrict__ agg, const float* __restrict__ xw,
                             const float* __restrict__ dinv, const float* __restrict__ bgcn,
                             float* __restrict__ h, float* __restrict__ g) {
  int idx = blockIdx.x * 256 + threadIdx.x;
  if (idx >= N_NODES * HID) return;
  int i = idx >> 7, j = idx & 127;
  float di = dinv[i];
  float v = leaky(agg[idx] + xw[idx] * di * di + bgcn[j]);
  h[idx] = v;
  g[idx] = v;
}

// ---------- fused GRU: gi/gh via WMMA into LDS, gates elementwise, g updated in place ----------
__global__ __launch_bounds__(256) void k_gru(const float* __restrict__ M, float* __restrict__ G,
                                             const __bf16* __restrict__ WihP, const __bf16* __restrict__ WhhP,
                                             const float* __restrict__ bih, const float* __restrict__ bhh) {
  __shared__ float sGi[16 * 384];
  __shared__ float sGh[16 * 384];
  int tile = blockIdx.x;
  int lane = threadIdx.x & 31;
  int wave = threadIdx.x >> 5;
  const float* Am = M + (size_t)tile * 16 * HID;
  const float* Ah = G + (size_t)tile * 16 * HID;

  // Hoist both A-operand fragment sets (messages + hidden state).
  bf16x16 am0 = load_a_frag(Am, HID, 0, lane);
  bf16x16 am1 = load_a_frag(Am, HID, 1, lane);
  bf16x16 am2 = load_a_frag(Am, HID, 2, lane);
  bf16x16 am3 = load_a_frag(Am, HID, 3, lane);
  bf16x16 ag0 = load_a_frag(Ah, HID, 0, lane);
  bf16x16 ag1 = load_a_frag(Ah, HID, 1, lane);
  bf16x16 ag2 = load_a_frag(Ah, HID, 2, lane);
  bf16x16 ag3 = load_a_frag(Ah, HID, 3, lane);

  for (int ct = wave; ct < 24; ct += 8) {       // 384/16 = 24 column tiles
    // Prefetch all 8 B fragments for this column tile.
    bf16x16 bi0 = load_b_frag(WihP, ct, 0, lane);
    bf16x16 bi1 = load_b_frag(WihP, ct, 1, lane);
    bf16x16 bi2 = load_b_frag(WihP, ct, 2, lane);
    bf16x16 bi3 = load_b_frag(WihP, ct, 3, lane);
    bf16x16 bh0 = load_b_frag(WhhP, ct, 0, lane);
    bf16x16 bh1 = load_b_frag(WhhP, ct, 1, lane);
    bf16x16 bh2 = load_b_frag(WhhP, ct, 2, lane);
    bf16x16 bh3 = load_b_frag(WhhP, ct, 3, lane);

    // 4 independent accumulators: dependent WMMAs are 4 apart -> hazard-free chain.
    f32x8 ai0 = {}, ai1 = {}, ah0 = {}, ah1 = {};
    ai0 = wmma_bf16(am0, bi0, ai0);
    ah0 = wmma_bf16(ag0, bh0, ah0);
    ai1 = wmma_bf16(am1, bi1, ai1);
    ah1 = wmma_bf16(ag1, bh1, ah1);
    ai0 = wmma_bf16(am2, bi2, ai0);
    ah0 = wmma_bf16(ag2, bh2, ah0);
    ai1 = wmma_bf16(am3, bi3, ai1);
    ah1 = wmma_bf16(ag3, bh3, ah1);

    float bi = bih[ct * 16 + (lane & 15)];
    float bh = bhh[ct * 16 + (lane & 15)];
    int n  = ct * 16 + (lane & 15);
    int mb = (lane & 16) ? 8 : 0;
#pragma unroll
    for (int r = 0; r < 8; ++r) {
      sGi[(mb + r) * 384 + n] = ai0[r] + ai1[r] + bi;
      sGh[(mb + r) * 384 + n] = ah0[r] + ah1[r] + bh;
    }
  }
  __syncthreads();

  for (int idx = threadIdx.x; idx < 16 * HID; idx += 256) {
    int m = idx >> 7, j = idx & 127;
    float ir = sGi[m * 384 + j],       hr = sGh[m * 384 + j];
    float iz = sGi[m * 384 + 128 + j], hz = sGh[m * 384 + 128 + j];
    float in = sGi[m * 384 + 256 + j], hn = sGh[m * 384 + 256 + j];
    float r = sigmoidf_(ir + hr);
    float z = sigmoidf_(iz + hz);
    float n = tanhf(in + r * hn);
    size_t gi = (size_t)(tile * 16 + m) * HID + j;
    float hp = G[gi];
    G[gi] = (1.f - z) * n + z * hp;
  }
}

__global__ void k_resid(const float* __restrict__ g, const float* __restrict__ h, float* __restrict__ t) {
  int idx = blockIdx.x * 256 + threadIdx.x;
  if (idx < N_NODES * HID) t[idx] = leaky(g[idx]) + h[idx];
}

__global__ void k_probs(const float* __restrict__ h2, const float* __restrict__ W2,
                        const float* __restrict__ b2, float* __restrict__ probs) {
  int i = blockIdx.x * 256 + threadIdx.x;
  if (i >= N_NODES) return;
  const float* p = h2 + (size_t)i * HID;
  float s = 0.f;
#pragma unroll
  for (int j = 0; j < HID; j += 4) {
    float4 a = *(const float4*)(p + j);
    float4 w = *(const float4*)(W2 + j);
    s += a.x * w.x + a.y * w.y + a.z * w.z + a.w * w.w;
  }
  probs[i] = sigmoidf_(s + b2[0]);
}

__global__ void k_loss_edges(const int* __restrict__ row, const int* __restrict__ col,
                             const int* __restrict__ batch, const float* __restrict__ probs,
                             float* __restrict__ accW) {
  int e = blockIdx.x * 256 + threadIdx.x;
  if (e >= N_EDGES) return;
  atomicAdd(&accW[batch[row[e]]], probs[row[e]] * probs[col[e]]);
}

__global__ void k_loss_nodes(const int* __restrict__ batch, const float* __restrict__ probs,
                             const float* __restrict__ degR, float* __restrict__ accD) {
  int i = blockIdx.x * 256 + threadIdx.x;
  if (i >= N_NODES) return;
  atomicAdd(&accD[batch[i]], probs[i] * degR[i]);
}

__global__ void k_out(const float* __restrict__ accW, const float* __restrict__ accD, float* __restrict__ out) {
  int g = threadIdx.x;
  if (g < NGRAPH) out[g] = -(accD[g] - accW[g]) * 0.5f;
}

extern "C" void kernel_launch(void* const* d_in, const int* in_sizes, int n_in,
                              void* d_out, int out_size, void* d_ws, size_t ws_size,
                              hipStream_t stream) {
  (void)in_sizes; (void)n_in; (void)out_size; (void)ws_size;
  const float* x     = (const float*)d_in[0];
  const int*   ei    = (const int*)  d_in[1];
  const int*   batch = (const int*)  d_in[2];
  const float* Wgcn  = (const float*)d_in[3];
  const float* bgcn  = (const float*)d_in[4];
  const float* Wgg   = (const float*)d_in[5];
  const float* Wih   = (const float*)d_in[6];
  const float* Whh   = (const float*)d_in[7];
  const float* bih   = (const float*)d_in[8];
  const float* bhh   = (const float*)d_in[9];
  const float* W1    = (const float*)d_in[10];
  const float* b1    = (const float*)d_in[11];
  const float* W2    = (const float*)d_in[12];
  const float* b2    = (const float*)d_in[13];
  const int* row = ei;
  const int* col = ei + N_EDGES;

  const int NH = N_NODES * HID;              // 6,400,000
  float* ws   = (float*)d_ws;
  float* xw   = ws;                          // x@Wgcn, later reused as t / hres
  float* agg  = ws + (size_t)NH;             // scatter accumulator (m)
  float* hbuf = ws + (size_t)2 * NH;
  float* gbuf = ws + (size_t)3 * NH;
  float* degC = ws + (size_t)4 * NH;
  float* degR = degC + N_NODES;
  float* dinv = degR + N_NODES;
  float* prob = dinv + N_NODES;
  float* accW = prob + N_NODES;              // [G]
  float* accD = accW + NGRAPH;               // [G]
  __bf16* WgcnP = (__bf16*)(accD + NGRAPH);  // 32B-aligned by construction
  __bf16* WggP  = WgcnP + 16384;             // 3 layers x 16384
  __bf16* W1P   = WggP + 3 * 16384;
  __bf16* WihP  = W1P + 16384;               // 128x384 packed
  __bf16* WhhP  = WihP + 49152;

  dim3 B(256);
  const int tiles = N_NODES / 16;            // 3125 exact

  // degrees + norm
  k_zero  <<<(2 * N_NODES + 255) / 256, B, 0, stream>>>(degC, 2 * N_NODES);
  k_degree<<<(N_EDGES + 255) / 256,     B, 0, stream>>>(row, col, degC, degR);
  k_dinv  <<<(N_NODES + 255) / 256,     B, 0, stream>>>(degC, dinv);

  // pack weights to WMMA fragment layout (bf16)
  k_pack<<<(16384 + 255) / 256, B, 0, stream>>>(Wgcn, WgcnP, 128, 0);
  for (int l = 0; l < 3; ++l)
    k_pack<<<(16384 + 255) / 256, B, 0, stream>>>(Wgg + l * 16384, WggP + l * 16384, 128, 0);
  k_pack<<<(16384 + 255) / 256, B, 0, stream>>>(W1, W1P, 128, 0);
  k_pack<<<(49152 + 255) / 256, B, 0, stream>>>(Wih, WihP, 384, 1);
  k_pack<<<(49152 + 255) / 256, B, 0, stream>>>(Whh, WhhP, 384, 1);

  // GCN layer
  k_gemm<<<tiles, B, 0, stream>>>(x, WgcnP, nullptr, xw, 0);
  k_zero<<<(NH + 255) / 256, B, 0, stream>>>(agg, NH);
  k_scatter<<<N_EDGES / 8, B, 0, stream>>>(xw, row, col, dinv, agg, 1);
  k_gcn_finish<<<(NH + 255) / 256, B, 0, stream>>>(agg, xw, dinv, bgcn, hbuf, gbuf);

  // 3x GatedGraphConv (message GEMM -> scatter -> fused GRU)
  for (int l = 0; l < 3; ++l) {
    k_gemm<<<tiles, B, 0, stream>>>(gbuf, WggP + l * 16384, nullptr, xw, 0);
    k_zero<<<(NH + 255) / 256, B, 0, stream>>>(agg, NH);
    k_scatter<<<N_EDGES / 8, B, 0, stream>>>(xw, row, col, dinv, agg, 0);
    k_gru<<<tiles, B, 0, stream>>>(agg, gbuf, WihP, WhhP, bih, bhh);
  }

  // head: hres = leaky(g)+h ; h2 = leaky(hres@W1+b1) ; probs = sigmoid(h2@W2+b2)
  k_resid<<<(NH + 255) / 256, B, 0, stream>>>(gbuf, hbuf, xw);
  k_gemm <<<tiles, B, 0, stream>>>(xw, W1P, b1, gbuf, 1);
  k_probs<<<(N_NODES + 255) / 256, B, 0, stream>>>(gbuf, W2, b2, prob);

  // maxcut loss per graph
  k_zero<<<1, B, 0, stream>>>(accW, 2 * NGRAPH);
  k_loss_edges<<<(N_EDGES + 255) / 256, B, 0, stream>>>(row, col, batch, prob, accW);
  k_loss_nodes<<<(N_NODES + 255) / 256, B, 0, stream>>>(batch, prob, degR, accD);
  k_out<<<1, 64, 0, stream>>>(accW, accD, (float*)d_out);
}
